// Multi_Image_Fusion_89481348644947
// MI455X (gfx1250) — matrix operand order repaired and checked by reading, not compiled, not run on hardware
//
#include <hip/hip_runtime.h>

typedef __attribute__((ext_vector_type(16))) _Float16 v16h;
typedef __attribute__((ext_vector_type(8)))  _Float16 v8h;
typedef __attribute__((ext_vector_type(8)))  float    v8f;

union Frag { v16h v; v8h h[2]; _Float16 e[16]; };

#define BATCH 2
#define KF 8
#define NQ 3072
#define NKV 2048
#define NH 8

// ---------------------------------------------------------------------------
// f32 -> f16 convert (weights)
__global__ void cvt_f32_f16(const float* __restrict__ s, _Float16* __restrict__ d, int n) {
    int i = blockIdx.x * blockDim.x + threadIdx.x;
    if (i < n) d[i] = (_Float16)s[i];
}

// (B, C, Ntok) f32 -> (B, Ntok, C) f16
__global__ void tconv_f32_f16(const float* __restrict__ s, _Float16* __restrict__ d,
                              int C, int Ntok, int total) {
    int i = blockIdx.x * blockDim.x + threadIdx.x;
    if (i >= total) return;
    int n = i % Ntok; int c = (i / Ntok) % C; int b = i / (Ntok * C);
    d[((size_t)b * Ntok + n) * C + c] = (_Float16)s[i];
}

// ---------------------------------------------------------------------------
// triplane coords -> per-frame image coords xy (B,K,NQ,2)
__global__ void proj_kernel(const float* __restrict__ proj_mat, float* __restrict__ xy) {
    int idx = blockIdx.x * blockDim.x + threadIdx.x;
    if (idx >= BATCH * KF * NQ) return;
    int q = idx % NQ; int kf = (idx / NQ) % KF; int b = idx / (NQ * KF);
    int hh = q >> 5;          // row in 0..95
    int w  = q & 31;
    int plane = hh >> 5; int i = hh & 31; int j = w;
    const float mid = 15.5f;  // (32-1)/2
    float cx, cy, cz;
    if (plane == 0)      { cx = (float)j; cy = mid;      cz = (float)i; }   // xz
    else if (plane == 1) { cx = (float)j; cy = (float)i; cz = mid; }        // xy
    else                 { cx = mid;      cy = (float)j; cz = (float)i; }   // yz
    const float sc = 2.0f * (1.0f + 0.1f + 1e-5f);
    cx = (cx / 31.0f - 0.5f) * sc;
    cy = (cy / 31.0f - 0.5f) * sc;
    cz = (cz / 31.0f - 0.5f) * sc;
    const float* P = proj_mat + (size_t)(b * KF + kf) * 16;
    float c0 = cx * P[0] + cy * P[1] + cz * P[2]  + P[3];
    float c1 = cx * P[4] + cy * P[5] + cz * P[6]  + P[7];
    float c2 = cx * P[8] + cy * P[9] + cz * P[10] + P[11];
    float fs = 15.0f / 223.0f;
    xy[(size_t)idx * 2 + 0] = c0 / c2 * fs;
    xy[(size_t)idx * 2 + 1] = c1 / c2 * fs;
}

// visibility bitmask: word (b, q, t) packs kv = 32t..32t+31 (bit=1 -> masked/-inf)
__global__ void mask_kernel(const float* __restrict__ xy, const float* __restrict__ valid_frames,
                            unsigned* __restrict__ maskbuf) {
    int idx = blockIdx.x * blockDim.x + threadIdx.x;
    if (idx >= BATCH * NQ * 64) return;
    int t = idx & 63; int q = (idx >> 6) % NQ; int b = idx / (NQ * 64);
    int kf = t >> 3;               // 256 px per frame, words of 32 align
    int pbase = (t & 7) * 32;
    size_t xo = ((size_t)(b * KF + kf) * NQ + q) * 2;
    float vx = xy[xo], vy = xy[xo + 1];
    bool invalid = (1.0f - valid_frames[b * KF + kf]) > 0.0f;
    unsigned wmask = 0u;
    #pragma unroll
    for (int i = 0; i < 32; ++i) {
        int p = pbase + i;
        float dx = vx - (float)(p & 15);
        float dy = vy - (float)(p >> 4);
        bool m = (dx * dx + dy * dy) < 4.0f;   // dist < SIGMA(2)
        wmask |= ((m | invalid) ? 1u : 0u) << i;
    }
    maskbuf[idx] = wmask;
}

// ---------------------------------------------------------------------------
// Generic WMMA GEMM: C[b,m,n] = sum_k A[b,m,k]*W[n,k] + bias[n], 16x64 per wave.
// Software-pipelined: fragments for k0+32 are loaded into a second buffer set
// before the 4 WMMAs on the current set issue, then buffers rotate. WMMAs thus
// depend only on older loads -> partial loadcnt waits, loads overlap XDL.
// outMode: 0 = f16 (b,M,N); 1 = f32 (b,N,M); 2 = f16 (b,N,M)
__global__ __launch_bounds__(32, 1)
void gemm16_wmma(const _Float16* __restrict__ A, const _Float16* __restrict__ W,
                 const float* __restrict__ bias, void* __restrict__ Cout,
                 int M, int N, int Kd, float scaleOut, int outMode) {
    const int lane = threadIdx.x & 31;
    const int hl = lane >> 4;
    const int lr = lane & 15;
    const int mbase = blockIdx.x * 16;
    const int nbase = blockIdx.y * 64;
    const int b = blockIdx.z;
    const _Float16* Ab = A + (size_t)b * M * Kd;

    v8f acc[4];
    #pragma unroll
    for (int t = 0; t < 4; ++t)
        #pragma unroll
        for (int e = 0; e < 8; ++e) acc[t][e] = 0.0f;

    // A fragment (16-bit A 16x32 layout: runs [8*hl,+8) and [16+8*hl,+8))
    // B fragment (32x16): lane = col n, elements K runs [8*hl,+8) and [16+8*hl,+8)
    const _Float16* pa_row = Ab + (size_t)(mbase + lr) * Kd + hl * 8;
    const _Float16* pw_row = W + (size_t)(nbase + lr) * Kd + hl * 8;

    Frag a0, b0[4];
    a0.h[0] = *(const v8h*)(pa_row);
    a0.h[1] = *(const v8h*)(pa_row + 16);
    #pragma unroll
    for (int t = 0; t < 4; ++t) {
        const _Float16* pw = pw_row + (size_t)t * 16 * Kd;
        b0[t].h[0] = *(const v8h*)pw;
        b0[t].h[1] = *(const v8h*)(pw + 16);
    }

    for (int k0 = 32; k0 < Kd; k0 += 32) {
        // prefetch next k-slice into the second buffer set
        Frag a1, b1[4];
        a1.h[0] = *(const v8h*)(pa_row + k0);
        a1.h[1] = *(const v8h*)(pa_row + k0 + 16);
        __builtin_prefetch(pa_row + k0 + 32, 0, 1);
        #pragma unroll
        for (int t = 0; t < 4; ++t) {
            const _Float16* pw = pw_row + (size_t)t * 16 * Kd + k0;
            b1[t].h[0] = *(const v8h*)pw;
            b1[t].h[1] = *(const v8h*)(pw + 16);
        }
        // compute on the current set (depends only on older loads)
        #pragma unroll
        for (int t = 0; t < 4; ++t)
            acc[t] = __builtin_amdgcn_wmma_f32_16x16x32_f16(
                false, a0.v, false, b0[t].v, (short)0, acc[t], false, false);
        // rotate
        a0 = a1;
        #pragma unroll
        for (int t = 0; t < 4; ++t) b0[t] = b1[t];
    }
    #pragma unroll
    for (int t = 0; t < 4; ++t)
        acc[t] = __builtin_amdgcn_wmma_f32_16x16x32_f16(
            false, a0.v, false, b0[t].v, (short)0, acc[t], false, false);

    #pragma unroll
    for (int t = 0; t < 4; ++t) {
        int n = nbase + t * 16 + lr;
        float bv = bias ? bias[n] : 0.0f;
        #pragma unroll
        for (int g = 0; g < 8; ++g) {
            int m = mbase + g + hl * 8;       // C layout: row = g + 8*(lane>>4)
            float val = (acc[t][g] + bv) * scaleOut;
            if (outMode == 0)
                ((_Float16*)Cout)[((size_t)b * M + m) * N + n] = (_Float16)val;
            else if (outMode == 1)
                ((float*)Cout)[((size_t)b * N + n) * M + m] = val;
            else
                ((_Float16*)Cout)[((size_t)b * N + n) * M + m] = (_Float16)val;
        }
    }
}

// ---------------------------------------------------------------------------
// Flash attention: one wave per (b, head, 16-query tile). d=16 padded to K=32.
__global__ __launch_bounds__(32, 1)
void attn_wmma(const _Float16* __restrict__ qhf,   // (B,NQ,128) f16, scale folded in
               const _Float16* __restrict__ khf,   // (B,NKV,128) f16
               const _Float16* __restrict__ vT,    // (B,128,NKV) f16 (d-major)
               const unsigned* __restrict__ maskbuf,
               _Float16* __restrict__ attnf) {     // (B,NQ,128) f16
    __shared__ _Float16 Plds[16 * 32];
    const int lane = threadIdx.x & 31;
    const int hl = lane >> 4;
    const int lr = lane & 15;
    const int qbase = blockIdx.x * 16;
    const int h = blockIdx.y;
    const int b = blockIdx.z;

    Frag aq;   // Q: row = query, K(d) padded: real d in elements 0..7 per half
    {
        const _Float16* p = qhf + ((size_t)(b * NQ + qbase + lr)) * 128 + h * 16 + hl * 8;
        aq.h[0] = *(const v8h*)p;
        #pragma unroll
        for (int e = 8; e < 16; ++e) aq.e[e] = (_Float16)0.0f;
    }

    v8f oacc;
    #pragma unroll
    for (int e = 0; e < 8; ++e) oacc[e] = 0.0f;
    float mrow[8], lrow[8];
    #pragma unroll
    for (int g = 0; g < 8; ++g) { mrow[g] = -__builtin_inff(); lrow[g] = 0.0f; }

    for (int t = 0; t < NKV / 32; ++t) {
        const int kvb = t * 32;
        Frag kb0, kb1;   // K^T B-frags: lane = kv col, elements 0..7 = d 8hl..8hl+7, 8..15 pad
        Frag vb;         // V B-frag:    lane = d col, elements = kv (loaded early)
        {
            const _Float16* p = khf + ((size_t)(b * NKV + kvb + lr)) * 128 + h * 16 + hl * 8;
            kb0.h[0] = *(const v8h*)p;
            p += 16 * 128;
            kb1.h[0] = *(const v8h*)p;
            #pragma unroll
            for (int e = 8; e < 16; ++e) { kb0.e[e] = (_Float16)0.0f; kb1.e[e] = (_Float16)0.0f; }
        }
        {
            const _Float16* p = vT + ((size_t)(b * 128 + h * 16 + lr)) * NKV + kvb + hl * 8;
            vb.h[0] = *(const v8h*)p;
            vb.h[1] = *(const v8h*)(p + 16);
        }
        v8f zc;
        #pragma unroll
        for (int e = 0; e < 8; ++e) zc[e] = 0.0f;
        v8f s0 = __builtin_amdgcn_wmma_f32_16x16x32_f16(false, aq.v, false, kb0.v, (short)0, zc, false, false);
        v8f s1 = __builtin_amdgcn_wmma_f32_16x16x32_f16(false, aq.v, false, kb1.v, (short)0, zc, false, false);

        float p0[8], p1[8];
        #pragma unroll
        for (int g = 0; g < 8; ++g) {
            unsigned mw = maskbuf[((size_t)(b * NQ + qbase + g + hl * 8)) * 64 + t];
            float e0 = ((mw >> lr) & 1u)        ? -__builtin_inff() : s0[g];
            float e1 = ((mw >> (16 + lr)) & 1u) ? -__builtin_inff() : s1[g];
            float mt = fmaxf(e0, e1);
            #pragma unroll
            for (int sh = 1; sh < 16; sh <<= 1) mt = fmaxf(mt, __shfl_xor(mt, sh, 32));
            float mn = fmaxf(mrow[g], mt);
            float fac = (mrow[g] > -1e30f) ? __expf(mrow[g] - mn) : 0.0f;
            float q0 = (e0 > -1e30f) ? __expf(e0 - mn) : 0.0f;
            float q1 = (e1 > -1e30f) ? __expf(e1 - mn) : 0.0f;
            float rs = q0 + q1;
            #pragma unroll
            for (int sh = 1; sh < 16; sh <<= 1) rs += __shfl_xor(rs, sh, 32);
            lrow[g] = lrow[g] * fac + rs;
            mrow[g] = mn;
            oacc[g] = oacc[g] * fac;
            p0[g] = q0; p1[g] = q1;
        }

        // C-layout -> A-layout transpose of P through LDS
        #pragma unroll
        for (int g = 0; g < 8; ++g) {
            const int row = g + hl * 8;
            Plds[row * 32 + lr]      = (_Float16)p0[g];
            Plds[row * 32 + 16 + lr] = (_Float16)p1[g];
        }
        asm volatile("s_wait_dscnt 0x0" ::: "memory");

        Frag ap;
        {
            const _Float16* p = &Plds[lr * 32 + hl * 8];   // A runs: [8hl,+8),[16+8hl,+8)
            ap.h[0] = *(const v8h*)p;
            ap.h[1] = *(const v8h*)(p + 16);
        }
        oacc = __builtin_amdgcn_wmma_f32_16x16x32_f16(false, ap.v, false, vb.v, (short)0, oacc, false, false);
    }

    #pragma unroll
    for (int g = 0; g < 8; ++g) {
        int q = qbase + g + hl * 8;
        float inv = (lrow[g] > 0.0f) ? (1.0f / lrow[g]) : 0.0f;
        attnf[((size_t)(b * NQ + q)) * 128 + h * 16 + lr] = (_Float16)(oacc[g] * inv);
    }
}

// ---------------------------------------------------------------------------
extern "C" void kernel_launch(void* const* d_in, const int* in_sizes, int n_in,
                              void* d_out, int out_size, void* d_ws, size_t ws_size,
                              hipStream_t stream) {
    (void)in_sizes; (void)n_in; (void)out_size; (void)ws_size;
    const float* triplane_feat = (const float*)d_in[0];
    const float* image_feat    = (const float*)d_in[1];
    const float* proj_mat      = (const float*)d_in[2];
    const float* valid_frames  = (const float*)d_in[3];
    const float* conv_w = (const float*)d_in[4];  const float* conv_b = (const float*)d_in[5];
    const float* q_w    = (const float*)d_in[6];
    const float* k_w    = (const float*)d_in[7];  const float* k_b    = (const float*)d_in[8];
    const float* v_w    = (const float*)d_in[9];  const float* v_b    = (const float*)d_in[10];
    const float* wq     = (const float*)d_in[11]; const float* wq_b   = (const float*)d_in[12];
    const float* wk     = (const float*)d_in[13]; const float* wk_b   = (const float*)d_in[14];
    const float* wv     = (const float*)d_in[15]; const float* wv_b   = (const float*)d_in[16];
    const float* ao_w   = (const float*)d_in[17]; const float* ao_b   = (const float*)d_in[18];
    const float* out_w  = (const float*)d_in[19]; const float* out_b  = (const float*)d_in[20];

    char* ws = (char*)d_ws;
    size_t off = 0;
    auto alloc = [&](size_t bytes) { size_t o = off; off += (bytes + 255) & ~(size_t)255; return o; };
    _Float16* Xh   = (_Float16*)(ws + alloc((size_t)BATCH * NKV * 1280 * 2));
    _Float16* Th   = (_Float16*)(ws + alloc((size_t)BATCH * NQ * 64 * 2));
    _Float16* Wc   = (_Float16*)(ws + alloc((size_t)512 * 1280 * 2));
    _Float16* Wk   = (_Float16*)(ws + alloc((size_t)128 * 512 * 2));
    _Float16* Wv   = (_Float16*)(ws + alloc((size_t)128 * 512 * 2));
    _Float16* Wqw  = (_Float16*)(ws + alloc((size_t)128 * 64 * 2));
    _Float16* Wwq  = (_Float16*)(ws + alloc((size_t)128 * 128 * 2));
    _Float16* Wwk  = (_Float16*)(ws + alloc((size_t)128 * 128 * 2));
    _Float16* Wwv  = (_Float16*)(ws + alloc((size_t)128 * 128 * 2));
    _Float16* Wao  = (_Float16*)(ws + alloc((size_t)128 * 128 * 2));
    _Float16* Wo   = (_Float16*)(ws + alloc((size_t)64 * 128 * 2));
    _Float16* imgh = (_Float16*)(ws + alloc((size_t)BATCH * NKV * 512 * 2));
    _Float16* kh1  = (_Float16*)(ws + alloc((size_t)BATCH * NKV * 128 * 2));
    _Float16* vh1  = (_Float16*)(ws + alloc((size_t)BATCH * NKV * 128 * 2));
    _Float16* khf  = (_Float16*)(ws + alloc((size_t)BATCH * NKV * 128 * 2));
    _Float16* vT   = (_Float16*)(ws + alloc((size_t)BATCH * 128 * NKV * 2));
    _Float16* qf   = (_Float16*)(ws + alloc((size_t)BATCH * NQ * 128 * 2));
    _Float16* qhf  = (_Float16*)(ws + alloc((size_t)BATCH * NQ * 128 * 2));
    float*    xy   = (float*)   (ws + alloc((size_t)BATCH * KF * NQ * 2 * 4));
    unsigned* mbuf = (unsigned*)(ws + alloc((size_t)BATCH * NQ * 64 * 4));
    _Float16* attnf= (_Float16*)(ws + alloc((size_t)BATCH * NQ * 128 * 2));
    _Float16* a2h  = (_Float16*)(ws + alloc((size_t)BATCH * NQ * 128 * 2));

    auto cvt = [&](const float* s, _Float16* d, int n) {
        cvt_f32_f16<<<(n + 255) / 256, 256, 0, stream>>>(s, d, n);
    };
    cvt(conv_w, Wc, 512 * 1280);
    cvt(k_w, Wk, 128 * 512);    cvt(v_w, Wv, 128 * 512);
    cvt(q_w, Wqw, 128 * 64);
    cvt(wq, Wwq, 128 * 128);    cvt(wk, Wwk, 128 * 128);    cvt(wv, Wwv, 128 * 128);
    cvt(ao_w, Wao, 128 * 128);  cvt(out_w, Wo, 64 * 128);

    { int tot = BATCH * 1280 * NKV;
      tconv_f32_f16<<<(tot + 255) / 256, 256, 0, stream>>>(image_feat, Xh, 1280, NKV, tot); }
    { int tot = BATCH * 64 * NQ;
      tconv_f32_f16<<<(tot + 255) / 256, 256, 0, stream>>>(triplane_feat, Th, 64, NQ, tot); }

    proj_kernel<<<(BATCH * KF * NQ + 255) / 256, 256, 0, stream>>>(proj_mat, xy);
    mask_kernel<<<(BATCH * NQ * 64 + 255) / 256, 256, 0, stream>>>(xy, valid_frames, mbuf);

    auto gemm = [&](const _Float16* A, const _Float16* W, const float* bias, void* C,
                    int M, int N, int Kd, float sc, int mode) {
        dim3 grid(M / 16, N / 64, BATCH);
        gemm16_wmma<<<grid, 32, 0, stream>>>(A, W, bias, C, M, N, Kd, sc, mode);
    };
    // conv 1x1: tokens(2048)x1280 -> 512
    gemm(Xh,   Wc,  conv_b, imgh, NKV, 512, 1280, 1.0f, 0);
    // k/v projections 512 -> 128
    gemm(imgh, Wk,  k_b,    kh1,  NKV, 128, 512,  1.0f, 0);
    gemm(imgh, Wv,  v_b,    vh1,  NKV, 128, 512,  1.0f, 0);
    // MHA internal projections 128 -> 128 (V written d-major for attention B-frags)
    gemm(kh1,  Wwk, wk_b,   khf,  NKV, 128, 128,  1.0f, 0);
    gemm(vh1,  Wwv, wv_b,   vT,   NKV, 128, 128,  1.0f, 2);
    // q path: 64 -> 128, then wq with 1/sqrt(d)=0.25 folded into epilogue
    gemm(Th,   Wqw, nullptr, qf,  NQ,  128, 64,   1.0f, 0);
    gemm(qf,   Wwq, wq_b,   qhf,  NQ,  128, 128,  0.25f, 0);

    { dim3 grid(NQ / 16, NH, BATCH);
      attn_wmma<<<grid, 32, 0, stream>>>(qhf, khf, vT, mbuf, attnf); }

    // output projections; final writes f32 transposed (B,64,3072) straight into d_out
    gemm(attnf, Wao, ao_b,  a2h,   NQ, 128, 128, 1.0f, 0);
    gemm(a2h,   Wo,  out_b, d_out, NQ, 64,  128, 1.0f, 1);
}
